// FusionGNN_4269197492590
// MI455X (gfx1250) — compile-verified, matrix operand order checked by reading, not run on hardware
//
#include <hip/hip_runtime.h>
#include <hip/hip_bf16.h>

typedef __attribute__((ext_vector_type(16))) _Float16 v16h;
typedef __attribute__((ext_vector_type(8)))  _Float16 v8h;
typedef __attribute__((ext_vector_type(8)))  float    v8f;

#define N_NODES  50000
#define N_EDGES  800000
#define N_GRAPHS 64
#define IN_CH    128
#define GLOBAL_DIM 256
#define HIDDEN   64
#define HEADS    4
#define N_CLASSES 10
#define QKV      (HEADS*HIDDEN)   // 256

// ---------------- utility: order-preserving float<->int for atomicMax ----------
__device__ __forceinline__ int fp_ord(float f) {
  int i = __float_as_int(f);
  return i >= 0 ? i : (i ^ 0x7fffffff);
}
__device__ __forceinline__ float fp_unord(int i) {
  return __int_as_float(i >= 0 ? i : (i ^ 0x7fffffff));
}

// ---------------- fill / convert / pack kernels --------------------------------
__global__ void k_fill_i32(int* __restrict__ p, int v, int n) {
  int t = blockIdx.x * blockDim.x + threadIdx.x;
  if (t < n) p[t] = v;
}
__global__ void k_fill_f32(float* __restrict__ p, float v, int n) {
  int t = blockIdx.x * blockDim.x + threadIdx.x;
  if (t < n) p[t] = v;
}
// f32 -> f16 bulk convert (A matrices, shared across Q/K/V/skip GEMMs)
__global__ void k_cvt_f16(const float* __restrict__ in, _Float16* __restrict__ out, int n) {
  int t = blockIdx.x * blockDim.x + threadIdx.x;
  if (t < n) out[t] = (_Float16)in[t];
}
// W[K,N] f32 -> Wt[N,K] f16 (column-major pack so B-fragment loads are contiguous)
__global__ void k_pack_wT(const float* __restrict__ W, _Float16* __restrict__ Wt, int K, int N) {
  int t = blockIdx.x * blockDim.x + threadIdx.x;
  if (t >= K * N) return;
  const int n = t / K, k = t - n * K;
  Wt[t] = (_Float16)W[(size_t)k * N + n];
}

// ---------------- WMMA GEMM: C[M,N] = relu?(A[M,K] @ W[K,N] + bias[N]) ---------
// One wave computes a 16(M) x 32(N) strip: 2 accumulators share one A fragment.
// K stepped by 32 with v_wmma_f32_16x16x32_f16; K is a multiple of 32 for all
// layers (128/64/256). Indices are CLAMPED (not predicated) so every fragment
// load is an unconditional contiguous 16-byte load; invalid columns are just
// never stored. Tile selection is wave-uniform -> EXEC all-ones at every WMMA.
__global__ void k_gemm_wmma(const _Float16* __restrict__ Ah, const _Float16* __restrict__ Wt,
                            const float* __restrict__ bias, float* __restrict__ C,
                            int M, int K, int N, int ldc, int relu)
{
  const int wave   = threadIdx.x >> 5;
  const int lane   = threadIdx.x & 31;
  const int tilesN = (N + 31) >> 5;             // 32-wide N strips
  const int tilesM = (M + 15) >> 4;
  const int tile   = blockIdx.x * (blockDim.x >> 5) + wave;
  if (tile >= tilesM * tilesN) return;          // uniform across the wave
  const int tm = tile / tilesN;
  const int tn = tile - tm * tilesN;

  const int half = lane >> 4;                   // 0: K 0..7,16..23   1: K 8..15,24..31
  const int l16  = lane & 15;
  int rowA = tm * 16 + l16;  if (rowA >= M) rowA = M - 1;   // clamp (M always %16==0 here)
  const int col0 = tn * 32 + l16;
  const int col1 = col0 + 16;
  const int c0 = col0 < N ? col0 : N - 1;       // clamped load columns
  const int c1 = col1 < N ? col1 : N - 1;
  const _Float16* Ar  = Ah + (size_t)rowA * K;
  const _Float16* B0r = Wt + (size_t)c0 * K;
  const _Float16* B1r = Wt + (size_t)c1 * K;

  v8f acc0 = {}, acc1 = {};
  for (int k0 = 0; k0 < K; k0 += 32) {
    const int kb = k0 + half * 8;
    const v8h alo  = *(const v8h*)(Ar  + kb);
    const v8h ahi  = *(const v8h*)(Ar  + kb + 16);
    const v8h b0lo = *(const v8h*)(B0r + kb);
    const v8h b0hi = *(const v8h*)(B0r + kb + 16);
    const v8h b1lo = *(const v8h*)(B1r + kb);
    const v8h b1hi = *(const v8h*)(B1r + kb + 16);
    const v16h a  = __builtin_shufflevector(alo,  ahi,  0,1,2,3,4,5,6,7,8,9,10,11,12,13,14,15);
    const v16h b0 = __builtin_shufflevector(b0lo, b0hi, 0,1,2,3,4,5,6,7,8,9,10,11,12,13,14,15);
    const v16h b1 = __builtin_shufflevector(b1lo, b1hi, 0,1,2,3,4,5,6,7,8,9,10,11,12,13,14,15);
    acc0 = __builtin_amdgcn_wmma_f32_16x16x32_f16(false, a, false, b0, (short)0, acc0, false, false);
    acc1 = __builtin_amdgcn_wmma_f32_16x16x32_f16(false, a, false, b1, (short)0, acc1, false, false);
  }

  // C layout: lane group half holds rows half*8 + r, column = lane%16 of the tile
#pragma unroll
  for (int r = 0; r < 8; ++r) {
    const int row = tm * 16 + half * 8 + r;
    if (row < M) {
      if (col0 < N) {
        float vv = acc0[r] + bias[col0];
        if (relu) vv = fmaxf(vv, 0.0f);
        C[(size_t)row * ldc + col0] = vv;
      }
      if (col1 < N) {
        float vv = acc1[r] + bias[col1];
        if (relu) vv = fmaxf(vv, 0.0f);
        C[(size_t)row * ldc + col1] = vv;
      }
    }
  }
}

// ---------------- edge pass 1: logits + segment max ----------------------------
__global__ void k_edge_logits(const float* __restrict__ q, const float* __restrict__ k,
                              const int* __restrict__ src, const int* __restrict__ dst,
                              float* __restrict__ s, int* __restrict__ m_i, int E)
{
  int t = blockIdx.x * blockDim.x + threadIdx.x;
  if (t >= E * HEADS) return;
  const int e = t >> 2, h = t & 3;
  const int sN = src[e], dN = dst[e];
  const float4* qp = (const float4*)(q + (size_t)dN * QKV + (h << 6));
  const float4* kp = (const float4*)(k + (size_t)sN * QKV + (h << 6));
  float acc = 0.0f;
#pragma unroll
  for (int i = 0; i < 16; ++i) {
    float4 a = qp[i], b = kp[i];
    acc += a.x * b.x + a.y * b.y + a.z * b.z + a.w * b.w;
  }
  acc *= 0.125f;                                // 1/sqrt(HIDDEN)
  s[(size_t)e * HEADS + h] = acc;
  atomicMax(&m_i[dN * HEADS + h], fp_ord(acc));
}

// ---------------- edge pass 2: exp-sum + weighted scatter-aggregate ------------
__global__ void k_edge_aggregate(const float* __restrict__ s, const int* __restrict__ m_i,
                                 const float* __restrict__ v,
                                 const int* __restrict__ src, const int* __restrict__ dst,
                                 float* __restrict__ z, float* __restrict__ agg, int E)
{
  int t = blockIdx.x * blockDim.x + threadIdx.x;
  if (t >= E * HEADS) return;
  const int e = t >> 2, h = t & 3;
  const int sN = src[e], dN = dst[e];
  const float mv = fp_unord(m_i[dN * HEADS + h]);
  const float ev = __expf(s[(size_t)e * HEADS + h] - mv);
  atomicAdd(&z[dN * HEADS + h], ev);
  const float* vp = v   + (size_t)sN * QKV + (h << 6);
  float*       ap = agg + (size_t)dN * QKV + (h << 6);
#pragma unroll 4
  for (int d = 0; d < HIDDEN; ++d) atomicAdd(&ap[d], ev * vp[d]);
}

// ---------------- node finalize: head-mean(agg/z) + skip, relu -----------------
__global__ void k_node_finalize(const float* __restrict__ agg, const float* __restrict__ z,
                                const float* __restrict__ skip, float* __restrict__ out, int N)
{
  int t = blockIdx.x * blockDim.x + threadIdx.x;
  if (t >= N * HIDDEN) return;
  const int n = t >> 6, d = t & 63;
  float acc = 0.0f;
#pragma unroll
  for (int h = 0; h < HEADS; ++h)
    acc += agg[(size_t)n * QKV + (h << 6) + d] / fmaxf(z[n * HEADS + h], 1e-30f);
  out[t] = fmaxf(acc * (1.0f / HEADS) + skip[t], 0.0f);
}

// ---------------- global mean pool ---------------------------------------------
__global__ void k_pool(const float* __restrict__ hfeat, const int* __restrict__ batch,
                       float* __restrict__ gsum, float* __restrict__ gcnt, int N)
{
  int t = blockIdx.x * blockDim.x + threadIdx.x;
  if (t >= N * HIDDEN) return;
  const int n = t >> 6, d = t & 63;
  const int g = batch[n];
  atomicAdd(&gsum[g * HIDDEN + d], hfeat[t]);
  if (d == 0) atomicAdd(&gcnt[g], 1.0f);
}
__global__ void k_pool_finalize(const float* __restrict__ gsum, const float* __restrict__ gcnt,
                                float* __restrict__ fused)
{
  int t = blockIdx.x * blockDim.x + threadIdx.x;
  if (t >= N_GRAPHS * HIDDEN) return;
  const int g = t >> 6, d = t & 63;
  fused[g * (2 * HIDDEN) + d] = gsum[t] / fmaxf(gcnt[g], 1.0f);
}

// ===============================================================================
extern "C" void kernel_launch(void* const* d_in, const int* in_sizes, int n_in,
                              void* d_out, int out_size, void* d_ws, size_t ws_size,
                              hipStream_t stream) {
  const float* x     = (const float*)d_in[0];
  const int*   eidx  = (const int*)  d_in[1];
  const int*   batch = (const int*)  d_in[2];
  const float* gfeat = (const float*)d_in[3];
  const float* c1_Wq = (const float*)d_in[4];  const float* c1_bq = (const float*)d_in[5];
  const float* c1_Wk = (const float*)d_in[6];  const float* c1_bk = (const float*)d_in[7];
  const float* c1_Wv = (const float*)d_in[8];  const float* c1_bv = (const float*)d_in[9];
  const float* c1_Ws = (const float*)d_in[10]; const float* c1_bs = (const float*)d_in[11];
  const float* c2_Wq = (const float*)d_in[12]; const float* c2_bq = (const float*)d_in[13];
  const float* c2_Wk = (const float*)d_in[14]; const float* c2_bk = (const float*)d_in[15];
  const float* c2_Wv = (const float*)d_in[16]; const float* c2_bv = (const float*)d_in[17];
  const float* c2_Ws = (const float*)d_in[18]; const float* c2_bs = (const float*)d_in[19];
  const float* g_W1  = (const float*)d_in[20]; const float* g_b1  = (const float*)d_in[21];
  const float* g_W2  = (const float*)d_in[22]; const float* g_b2  = (const float*)d_in[23];
  const float* h_W1  = (const float*)d_in[24]; const float* h_b1  = (const float*)d_in[25];
  const float* h_W2  = (const float*)d_in[26]; const float* h_b2  = (const float*)d_in[27];
  const int* src = eidx;
  const int* dst = eidx + N_EDGES;
  float* out = (float*)d_out;

  // ---- workspace layout (f32 units, 256-element aligned) ----
  float* ws = (float*)d_ws;
  size_t off = 0;
  auto alloc = [&](size_t n) { float* p = ws + off; off += (n + 255) & ~(size_t)255; return p; };
  float* q    = alloc((size_t)N_NODES * QKV);
  float* kbuf = alloc((size_t)N_NODES * QKV);
  float* v    = alloc((size_t)N_NODES * QKV);
  float* skip = alloc((size_t)N_NODES * HIDDEN);
  float* sbuf = alloc((size_t)N_EDGES * HEADS);
  int*   m_i  = (int*)alloc((size_t)N_NODES * HEADS);
  float* z    = alloc((size_t)N_NODES * HEADS);
  float* agg  = alloc((size_t)N_NODES * QKV);
  float* h1   = alloc((size_t)N_NODES * HIDDEN);
  float* h2   = alloc((size_t)N_NODES * HIDDEN);
  float* gsum = alloc(N_GRAPHS * HIDDEN);
  float* gcnt = alloc(N_GRAPHS);
  float* g1   = alloc(N_GRAPHS * HIDDEN);
  float* fused= alloc(N_GRAPHS * 2 * HIDDEN);
  float* hid  = alloc(N_GRAPHS * HIDDEN);
  // f16 staging (sizes in f32 units = half the element count)
  _Float16* xh    = (_Float16*)alloc((size_t)N_NODES * IN_CH  / 2);
  _Float16* h1h   = (_Float16*)alloc((size_t)N_NODES * HIDDEN / 2);
  _Float16* smallh= (_Float16*)alloc((size_t)N_GRAPHS * GLOBAL_DIM / 2);   // gfeat/g1/fused/hid staging
  _Float16* wt    = (_Float16*)alloc((size_t)GLOBAL_DIM * QKV / 2);        // packed W^T (max 256*256)
  (void)ws_size; (void)in_sizes; (void)n_in; (void)out_size;

  const int TB = 256;
  auto blocks = [](long long n, int tb) { return (int)((n + tb - 1) / tb); };
  auto cvt = [&](const float* src_, _Float16* dst_, long long n) {
    k_cvt_f16<<<blocks(n, TB), TB, 0, stream>>>(src_, dst_, (int)n);
  };
  // GEMM launcher: pack W^T to f16, then 4 waves/block, 16x32 strip per wave
  auto gemm = [&](const _Float16* Ah, const float* W, const float* b, float* C,
                  int M, int K, int N, int ldc, int relu) {
    k_pack_wT<<<blocks((long long)K * N, TB), TB, 0, stream>>>(W, wt, K, N);
    int tiles = ((M + 15) >> 4) * ((N + 31) >> 5);
    k_gemm_wmma<<<(tiles + 3) / 4, 128, 0, stream>>>(Ah, wt, b, C, M, K, N, ldc, relu);
  };
  // one TransformerConv layer (Ah = f16 input features)
  auto tconv = [&](const _Float16* Ah, int K,
                   const float* Wq, const float* bq, const float* Wk, const float* bk,
                   const float* Wv, const float* bv, const float* Wsk, const float* bsk,
                   float* hout) {
    gemm(Ah, Wq, bq, q,    N_NODES, K, QKV,    QKV,    0);
    gemm(Ah, Wk, bk, kbuf, N_NODES, K, QKV,    QKV,    0);
    gemm(Ah, Wv, bv, v,    N_NODES, K, QKV,    QKV,    0);
    gemm(Ah, Wsk, bsk, skip, N_NODES, K, HIDDEN, HIDDEN, 0);
    k_fill_i32<<<blocks(N_NODES * HEADS, TB), TB, 0, stream>>>(m_i, (int)0x80000000, N_NODES * HEADS);
    k_fill_f32<<<blocks(N_NODES * HEADS, TB), TB, 0, stream>>>(z, 0.0f, N_NODES * HEADS);
    k_fill_f32<<<blocks((long long)N_NODES * QKV, TB), TB, 0, stream>>>(agg, 0.0f, N_NODES * QKV);
    k_edge_logits<<<blocks((long long)N_EDGES * HEADS, TB), TB, 0, stream>>>(
        q, kbuf, src, dst, sbuf, m_i, N_EDGES);
    k_edge_aggregate<<<blocks((long long)N_EDGES * HEADS, TB), TB, 0, stream>>>(
        sbuf, m_i, v, src, dst, z, agg, N_EDGES);
    k_node_finalize<<<blocks((long long)N_NODES * HIDDEN, TB), TB, 0, stream>>>(
        agg, z, skip, hout, N_NODES);
  };

  // conv1 (K=128) -> h1, conv2 (K=64) -> h2
  cvt(x, xh, (long long)N_NODES * IN_CH);
  tconv(xh,  IN_CH,  c1_Wq, c1_bq, c1_Wk, c1_bk, c1_Wv, c1_bv, c1_Ws, c1_bs, h1);
  cvt(h1, h1h, (long long)N_NODES * HIDDEN);
  tconv(h1h, HIDDEN, c2_Wq, c2_bq, c2_Wk, c2_bk, c2_Wv, c2_bv, c2_Ws, c2_bs, h2);

  // global mean pool -> fused[:, 0:64]
  k_fill_f32<<<blocks(N_GRAPHS * HIDDEN, TB), TB, 0, stream>>>(gsum, 0.0f, N_GRAPHS * HIDDEN);
  k_fill_f32<<<1, 64, 0, stream>>>(gcnt, 0.0f, N_GRAPHS);
  k_pool<<<blocks((long long)N_NODES * HIDDEN, TB), TB, 0, stream>>>(h2, batch, gsum, gcnt, N_NODES);
  k_pool_finalize<<<blocks(N_GRAPHS * HIDDEN, TB), TB, 0, stream>>>(gsum, gcnt, fused);

  // global MLP: g2 written directly into fused[:, 64:128] via ldc = 128
  cvt(gfeat, smallh, (long long)N_GRAPHS * GLOBAL_DIM);
  gemm(smallh, g_W1, g_b1, g1, N_GRAPHS, GLOBAL_DIM, HIDDEN, HIDDEN, 1);
  cvt(g1, smallh, (long long)N_GRAPHS * HIDDEN);
  gemm(smallh, g_W2, g_b2, fused + HIDDEN, N_GRAPHS, HIDDEN, HIDDEN, 2 * HIDDEN, 1);

  // head: relu(fused @ h_W1 + h_b1) @ h_W2 + h_b2 -> d_out [64,10]
  cvt(fused, smallh, (long long)N_GRAPHS * 2 * HIDDEN);
  gemm(smallh, h_W1, h_b1, hid, N_GRAPHS, 2 * HIDDEN, HIDDEN, HIDDEN, 1);
  cvt(hid, smallh, (long long)N_GRAPHS * HIDDEN);
  gemm(smallh, h_W2, h_b2, out, N_GRAPHS, HIDDEN, N_CLASSES, N_CLASSES, 0);
}
